// SpatialAwareTrans_3341484556982
// MI455X (gfx1250) — compile-verified, matrix operand order checked by reading, not run on hardware
//
#include <hip/hip_runtime.h>

typedef __attribute__((ext_vector_type(16))) _Float16 v16h;
typedef __attribute__((ext_vector_type(8)))  _Float16 v8h;
typedef __attribute__((ext_vector_type(8)))  float    v8f;

union U16 { v16h v; v8h h[2]; };

#define NTOK 133280

__device__ __forceinline__ float wred_sum(float v) {
  for (int o = 16; o; o >>= 1) v += __shfl_xor(v, o);
  return v;
}
__device__ __forceinline__ float wred_max(float v) {
  for (int o = 16; o; o >>= 1) v = fmaxf(v, __shfl_xor(v, o));
  return v;
}
__device__ __forceinline__ float gelu_f(float x) {
  return 0.5f * x * (1.0f + erff(x * 0.70710678118654752440f));
}

// ---------------------------------------------------------------------------
// f32 -> f16 weight conversion
// ---------------------------------------------------------------------------
__global__ void f32_to_f16_kernel(const float* __restrict__ src,
                                  _Float16* __restrict__ dst, int n) {
  int i = blockIdx.x * 256 + threadIdx.x;
  if (i < n) dst[i] = (_Float16)src[i];
}

// ---------------------------------------------------------------------------
// Generic token GEMM: out = act(A[MxK] @ B[KxN] + bias) (+resid), f16 A/B.
// Block = 256 threads (8 waves); tile 128(M) x 64(N); K-step 32 (WMMA f16).
// A tile staged with CDNA5 async global->LDS copies (ASYNCcnt path).
// ---------------------------------------------------------------------------
__global__ __launch_bounds__(256) void gemm_tok(
    const _Float16* __restrict__ A, const _Float16* __restrict__ B,
    const float* __restrict__ bias, const float* __restrict__ resid,
    float* __restrict__ outF, _Float16* __restrict__ outH,
    int M, int N, int K, int act) {
  __shared__ _Float16 sA[128][40];
  __shared__ _Float16 sB[64][40];
  const int tid = threadIdx.x;
  const int m0 = blockIdx.y * 128;
  const int n0 = blockIdx.x * 64;
  const int lane = tid & 31, wv = tid >> 5;
  const int wm = (wv & 3) * 32, wn = (wv >> 2) * 32;
  const int half = lane >> 4, lr = lane & 15, kb = half * 8;

  v8f acc[2][2];
  for (int i = 0; i < 2; ++i)
    for (int j = 0; j < 2; ++j)
      for (int r = 0; r < 8; ++r) acc[i][j][r] = 0.0f;

  for (int k0 = 0; k0 < K; k0 += 32) {
    // A tile: 128x32 halves; 512 16-byte chunks moved straight into LDS
    // via async copies. OOB rows clamp to row M-1 (finite data; their C
    // rows are never stored).
    for (int it = 0; it < 2; ++it) {
      int chunk = tid + it * 256;
      int m = chunk >> 2;
      int kk = (chunk & 3) * 8;
      int gm = m0 + m;
      if (gm >= M) gm = M - 1;
      const _Float16* gp = A + (size_t)gm * K + k0 + kk;
      unsigned lds = (unsigned)(size_t)&sA[m][kk];
      asm volatile("global_load_async_to_lds_b128 %0, %1, off"
                   :: "v"(lds), "v"(gp)
                   : "memory");
    }
    // B tile transposed: sB[n][k] = B[(k0+k)*N + n0+n]
    for (int it = 0; it < 8; ++it) {
      int idx = tid + it * 256;
      int n = idx & 63;
      int k = idx >> 6;
      sB[n][k] = B[(size_t)(k0 + k) * N + n0 + n];
    }
    asm volatile("s_wait_asynccnt 0x0" ::: "memory");
    __syncthreads();
    U16 af[2], bf[2];
    for (int f = 0; f < 2; ++f) {
      int row = wm + f * 16 + lr;
      af[f].h[0] = *(const v8h*)&sA[row][kb];
      af[f].h[1] = *(const v8h*)&sA[row][kb + 16];
      int col = wn + f * 16 + lr;
      bf[f].h[0] = *(const v8h*)&sB[col][kb];
      bf[f].h[1] = *(const v8h*)&sB[col][kb + 16];
    }
    for (int i = 0; i < 2; ++i)
      for (int j = 0; j < 2; ++j)
        acc[i][j] = __builtin_amdgcn_wmma_f32_16x16x32_f16(
            false, af[i].v, false, bf[j].v, (short)0, acc[i][j], false, false);
    __syncthreads();
  }

  for (int i = 0; i < 2; ++i)
    for (int j = 0; j < 2; ++j)
      for (int r = 0; r < 8; ++r) {
        int gm = m0 + wm + i * 16 + half * 8 + r;
        int gn = n0 + wn + j * 16 + lr;
        if (gm < M) {
          float v = acc[i][j][r] + bias[gn];
          if (act) v = gelu_f(v);
          size_t o = (size_t)gm * N + gn;
          if (resid) v += resid[o];
          if (outF) outF[o] = v;
          else outH[o] = (_Float16)v;
        }
      }
}

// ---------------------------------------------------------------------------
// Stage-1 fc GEMM: A gathered from x (B,C,H,W) f32; scatter to windowed tokens.
// N fixed = 256, K = C.
// ---------------------------------------------------------------------------
__global__ __launch_bounds__(256) void gemm_fc(
    const float* __restrict__ x, const _Float16* __restrict__ W,
    const float* __restrict__ bias, float* __restrict__ Xtok,
    int C, int HW, int Wd, int ww, int toff, int P) {
  __shared__ _Float16 sA[128][40];
  __shared__ _Float16 sB[64][40];
  const int tid = threadIdx.x;
  const int m0 = blockIdx.y * 128;
  const int n0 = blockIdx.x * 64;
  const int lane = tid & 31, wv = tid >> 5;
  const int wm = (wv & 3) * 32, wn = (wv >> 2) * 32;
  const int half = lane >> 4, lr = lane & 15, kb = half * 8;

  v8f acc[2][2];
  for (int i = 0; i < 2; ++i)
    for (int j = 0; j < 2; ++j)
      for (int r = 0; r < 8; ++r) acc[i][j][r] = 0.0f;

  for (int k0 = 0; k0 < C; k0 += 32) {
    for (int it = 0; it < 16; ++it) {
      int idx = tid + it * 256;   // 4096 elements
      int m = idx & 127;
      int k = idx >> 7;
      int p = m0 + m;
      _Float16 v = (_Float16)0.0f;
      if (p < P) {
        int b = p / HW, hw = p - b * HW;
        v = (_Float16)x[((size_t)b * C + (k0 + k)) * HW + hw];
      }
      sA[m][k] = v;
    }
    for (int it = 0; it < 8; ++it) {
      int idx = tid + it * 256;
      int n = idx & 63;
      int k = idx >> 6;
      sB[n][k] = W[(size_t)(k0 + k) * 256 + n0 + n];
    }
    __syncthreads();
    U16 af[2], bf[2];
    for (int f = 0; f < 2; ++f) {
      int row = wm + f * 16 + lr;
      af[f].h[0] = *(const v8h*)&sA[row][kb];
      af[f].h[1] = *(const v8h*)&sA[row][kb + 16];
      int col = wn + f * 16 + lr;
      bf[f].h[0] = *(const v8h*)&sB[col][kb];
      bf[f].h[1] = *(const v8h*)&sB[col][kb + 16];
    }
    for (int i = 0; i < 2; ++i)
      for (int j = 0; j < 2; ++j)
        acc[i][j] = __builtin_amdgcn_wmma_f32_16x16x32_f16(
            false, af[i].v, false, bf[j].v, (short)0, acc[i][j], false, false);
    __syncthreads();
  }

  for (int i = 0; i < 2; ++i)
    for (int j = 0; j < 2; ++j)
      for (int r = 0; r < 8; ++r) {
        int p = m0 + wm + i * 16 + half * 8 + r;
        int gn = n0 + wn + j * 16 + lr;
        if (p < P) {
          int b = p / HW, hw = p - b * HW;
          int h = hw / Wd, w2 = hw - h * Wd;
          int wi = h / ww, wj2 = w2 / ww;
          int tok = toff + (h - wi * ww) * ww + (w2 - wj2 * ww);
          int trow = ((b * 14 + wi) * 14 + wj2) * 85 + tok;
          Xtok[(size_t)trow * 256 + gn] = acc[i][j][r] + bias[gn];
        }
      }
}

// ---------------------------------------------------------------------------
// Stage-3 rev GEMM: A gathered from windowed tokens (f32 X); out (B,C,H,W).
// K fixed = 256, N = C.
// ---------------------------------------------------------------------------
__global__ __launch_bounds__(256) void gemm_rev(
    const float* __restrict__ Xtok, const _Float16* __restrict__ W,
    const float* __restrict__ bias, float* __restrict__ out,
    int C, int HW, int Wd, int ww, int toff, int P) {
  __shared__ _Float16 sA[128][40];
  __shared__ _Float16 sB[64][40];
  const int tid = threadIdx.x;
  const int m0 = blockIdx.y * 128;
  const int n0 = blockIdx.x * 64;
  const int lane = tid & 31, wv = tid >> 5;
  const int wm = (wv & 3) * 32, wn = (wv >> 2) * 32;
  const int half = lane >> 4, lr = lane & 15, kb = half * 8;

  v8f acc[2][2];
  for (int i = 0; i < 2; ++i)
    for (int j = 0; j < 2; ++j)
      for (int r = 0; r < 8; ++r) acc[i][j][r] = 0.0f;

  for (int k0 = 0; k0 < 256; k0 += 32) {
    for (int it = 0; it < 16; ++it) {
      int idx = tid + it * 256;
      int k = idx & 31;
      int m = idx >> 5;
      int p = m0 + m;
      _Float16 v = (_Float16)0.0f;
      if (p < P) {
        int b = p / HW, hw = p - b * HW;
        int h = hw / Wd, w2 = hw - h * Wd;
        int wi = h / ww, wj2 = w2 / ww;
        int tok = toff + (h - wi * ww) * ww + (w2 - wj2 * ww);
        int trow = ((b * 14 + wi) * 14 + wj2) * 85 + tok;
        v = (_Float16)Xtok[(size_t)trow * 256 + k0 + k];
      }
      sA[m][k] = v;
    }
    for (int it = 0; it < 8; ++it) {
      int idx = tid + it * 256;
      int n = idx & 63;
      int k = idx >> 6;
      sB[n][k] = W[(size_t)(k0 + k) * C + n0 + n];
    }
    __syncthreads();
    U16 af[2], bf[2];
    for (int f = 0; f < 2; ++f) {
      int row = wm + f * 16 + lr;
      af[f].h[0] = *(const v8h*)&sA[row][kb];
      af[f].h[1] = *(const v8h*)&sA[row][kb + 16];
      int col = wn + f * 16 + lr;
      bf[f].h[0] = *(const v8h*)&sB[col][kb];
      bf[f].h[1] = *(const v8h*)&sB[col][kb + 16];
    }
    for (int i = 0; i < 2; ++i)
      for (int j = 0; j < 2; ++j)
        acc[i][j] = __builtin_amdgcn_wmma_f32_16x16x32_f16(
            false, af[i].v, false, bf[j].v, (short)0, acc[i][j], false, false);
    __syncthreads();
  }

  for (int i = 0; i < 2; ++i)
    for (int j = 0; j < 2; ++j)
      for (int r = 0; r < 8; ++r) {
        int p = m0 + wm + i * 16 + half * 8 + r;
        int gn = n0 + wn + j * 16 + lr;
        if (p < P) {
          int b = p / HW, hw = p - b * HW;
          out[((size_t)b * C + gn) * HW + hw] = acc[i][j][r] + bias[gn];
        }
      }
}

// ---------------------------------------------------------------------------
// LayerNorm over 256 dims: one wave per token, 8 tokens per 256-thread block.
// Reads f32 X, writes f16.
// ---------------------------------------------------------------------------
__global__ __launch_bounds__(256) void ln_kernel(
    const float* __restrict__ X, const float* __restrict__ g,
    const float* __restrict__ b, _Float16* __restrict__ out, int NT) {
  int tid = threadIdx.x;
  int t = blockIdx.x * 8 + (tid >> 5);
  int lane = tid & 31;
  if (t >= NT) return;
  const float* xp = X + (size_t)t * 256 + lane * 8;
  float v[8], s = 0.0f, s2 = 0.0f;
  for (int i = 0; i < 8; ++i) {
    v[i] = xp[i];
    s += v[i];
    s2 += v[i] * v[i];
  }
  s = wred_sum(s);
  s2 = wred_sum(s2);
  float mu = s * (1.0f / 256.0f);
  float var = s2 * (1.0f / 256.0f) - mu * mu;
  float rs = rsqrtf(var + 1e-6f);
  _Float16* op = out + (size_t)t * 256 + lane * 8;
  for (int i = 0; i < 8; ++i) {
    int c = lane * 8 + i;
    op[i] = (_Float16)((v[i] - mu) * rs * g[c] + b[c]);
  }
}

// ---------------------------------------------------------------------------
// Windowed attention: one block per (window, head); 85 tokens, hd=32,
// NO 1/sqrt(d) scaling (matches reference). One wave per query row.
// qkv layout per token (768): [q|k|v] each 256, head*32 within.
// ---------------------------------------------------------------------------
__global__ __launch_bounds__(256) void attn_kernel(
    const _Float16* __restrict__ qkv, _Float16* __restrict__ out) {
  __shared__ float Qs[85][33];
  __shared__ float Ks[85][33];
  __shared__ float Vs[85][33];
  __shared__ float Ps[8][96];
  int tid = threadIdx.x;
  int bw = blockIdx.x >> 3;
  int head = blockIdx.x & 7;
  int tbase = bw * 85;

  for (int idx = tid; idx < 85 * 32; idx += 256) {
    int t = idx >> 5, d = idx & 31;
    size_t base = (size_t)(tbase + t) * 768 + head * 32 + d;
    Qs[t][d] = (float)qkv[base];
    Ks[t][d] = (float)qkv[base + 256];
    Vs[t][d] = (float)qkv[base + 512];
  }
  __syncthreads();

  int wv = tid >> 5, lane = tid & 31;
  for (int i = wv; i < 85; i += 8) {
    float s[3], mloc = -1e30f;
    for (int jj = 0; jj < 3; ++jj) {
      int j = lane + jj * 32;
      if (j < 85) {
        float d0 = 0.0f;
        for (int d = 0; d < 32; ++d) d0 += Qs[i][d] * Ks[j][d];
        s[jj] = d0;
        mloc = fmaxf(mloc, d0);
      } else {
        s[jj] = -1e30f;
      }
    }
    float M = wred_max(mloc);
    float e[3], ssum = 0.0f;
    for (int jj = 0; jj < 3; ++jj) {
      int j = lane + jj * 32;
      e[jj] = (j < 85) ? expf(s[jj] - M) : 0.0f;
      ssum += e[jj];
    }
    float S = wred_sum(ssum);
    for (int jj = 0; jj < 3; ++jj) Ps[wv][lane + jj * 32] = e[jj];
    float inv = 1.0f / S;
    float a = 0.0f;
    for (int j = 0; j < 85; ++j) a += Ps[wv][j] * Vs[j][lane];
    out[(size_t)(tbase + i) * 256 + head * 32 + lane] = (_Float16)(a * inv);
  }
}

// ---------------------------------------------------------------------------
// Host launcher
// ---------------------------------------------------------------------------
extern "C" void kernel_launch(void* const* d_in, const int* in_sizes, int n_in,
                              void* d_out, int out_size, void* d_ws,
                              size_t ws_size, hipStream_t stream) {
  (void)in_sizes; (void)n_in; (void)out_size; (void)ws_size;

  const float* xin[4] = {(const float*)d_in[0], (const float*)d_in[1],
                         (const float*)d_in[2], (const float*)d_in[3]};
  const float* fc_b[4] = {(const float*)d_in[5], (const float*)d_in[7],
                          (const float*)d_in[9], (const float*)d_in[11]};
  const float* ln1_g = (const float*)d_in[12];
  const float* ln1_b = (const float*)d_in[13];
  const float* qkv_b = (const float*)d_in[15];
  const float* proj_b = (const float*)d_in[17];
  const float* ln2_g = (const float*)d_in[18];
  const float* ln2_b = (const float*)d_in[19];
  const float* mlp_b1 = (const float*)d_in[21];
  const float* mlp_b2 = (const float*)d_in[23];
  const float* rev_b[4] = {(const float*)d_in[25], (const float*)d_in[27],
                           (const float*)d_in[29], (const float*)d_in[31]};

  // f16 weight offsets (in halves)
  const size_t OFF_FCW[4] = {0, 65536, 196608, 458752};
  const size_t OFF_QKV = 720896, OFF_PROJ = 917504;
  const size_t OFF_M1 = 983040, OFF_M2 = 1245184;
  const size_t OFF_REV[4] = {1507328, 1572864, 1703936, 1966080};

  char* ws = (char*)d_ws;
  _Float16* W16 = (_Float16*)ws;                       // 4,456,448 B
  float* X = (float*)(ws + 4456448);                   // 136,478,720 B
  _Float16* H16 = (_Float16*)(ws + 140935168);         // 68,239,360 B
  _Float16* BIG = (_Float16*)(ws + 209174528);         // 272,957,440 B

  // --- convert weights to f16 ---
  const int widx[12] = {4, 6, 8, 10, 14, 16, 20, 22, 24, 26, 28, 30};
  const size_t woff[12] = {OFF_FCW[0], OFF_FCW[1], OFF_FCW[2], OFF_FCW[3],
                           OFF_QKV, OFF_PROJ, OFF_M1, OFF_M2,
                           OFF_REV[0], OFF_REV[1], OFF_REV[2], OFF_REV[3]};
  const int wn[12] = {65536, 131072, 262144, 262144, 196608, 65536,
                      262144, 262144, 65536, 131072, 262144, 262144};
  for (int i = 0; i < 12; ++i)
    f32_to_f16_kernel<<<(wn[i] + 255) / 256, 256, 0, stream>>>(
        (const float*)d_in[widx[i]], W16 + woff[i], wn[i]);

  const int Cj[4] = {256, 512, 1024, 1024};
  const int HWj[4] = {12544, 3136, 784, 196};
  const int Wdj[4] = {112, 56, 28, 14};
  const int wwj[4] = {8, 4, 2, 1};
  const int toff[4] = {0, 64, 80, 84};
  const int Pj[4] = {100352, 25088, 6272, 1568};

  // --- stage 1: fc + windowing into X ---
  for (int j = 0; j < 4; ++j) {
    dim3 g(4, (Pj[j] + 127) / 128);
    gemm_fc<<<g, 256, 0, stream>>>(xin[j], W16 + OFF_FCW[j], fc_b[j], X,
                                   Cj[j], HWj[j], Wdj[j], wwj[j], toff[j],
                                   Pj[j]);
  }

  const int NT = NTOK;
  const int MBLK = (NT + 127) / 128;

  // --- LN1 -> H16 ---
  ln_kernel<<<(NT + 7) / 8, 256, 0, stream>>>(X, ln1_g, ln1_b, H16, NT);

  // --- QKV: H16 @ qkv_w -> BIG (f16, NTx768) ---
  {
    dim3 g(768 / 64, MBLK);
    gemm_tok<<<g, 256, 0, stream>>>(H16, W16 + OFF_QKV, qkv_b, nullptr,
                                    nullptr, BIG, NT, 768, 256, 0);
  }

  // --- attention -> H16 (f16, NTx256) ---
  attn_kernel<<<8 * 196 * 8, 256, 0, stream>>>(BIG, H16);

  // --- proj (+residual X) -> X ---
  {
    dim3 g(4, MBLK);
    gemm_tok<<<g, 256, 0, stream>>>(H16, W16 + OFF_PROJ, proj_b, X, X,
                                    nullptr, NT, 256, 256, 0);
  }

  // --- LN2 -> H16 ---
  ln_kernel<<<(NT + 7) / 8, 256, 0, stream>>>(X, ln2_g, ln2_b, H16, NT);

  // --- MLP1 (GELU) -> BIG (f16, NTx1024) ---
  {
    dim3 g(1024 / 64, MBLK);
    gemm_tok<<<g, 256, 0, stream>>>(H16, W16 + OFF_M1, mlp_b1, nullptr,
                                    nullptr, BIG, NT, 1024, 256, 1);
  }

  // --- MLP2 (+residual X) -> X ---
  {
    dim3 g(4, MBLK);
    gemm_tok<<<g, 256, 0, stream>>>(BIG, W16 + OFF_M2, mlp_b2, X, X, nullptr,
                                    NT, 256, 1024, 0);
  }

  // --- stage 3: rev GEMMs -> d_out (concatenated outputs) ---
  float* out = (float*)d_out;
  const size_t ooff[4] = {0, 25690112, 38535168, 44957696};
  for (int j = 0; j < 4; ++j) {
    dim3 g(Cj[j] / 64, (Pj[j] + 127) / 128);
    gemm_rev<<<g, 256, 0, stream>>>(X, W16 + OFF_REV[j], rev_b[j],
                                    out + ooff[j], Cj[j], HWj[j], Wdj[j],
                                    wwj[j], toff[j], Pj[j]);
  }
}
